// Predictor_13219909337541
// MI455X (gfx1250) — matrix operand order repaired and checked by reading, not compiled, not run on hardware
//
#include <hip/hip_runtime.h>

typedef __attribute__((ext_vector_type(16))) __bf16 v16bf;
typedef __attribute__((ext_vector_type(8)))  __bf16 v8bf;
typedef __attribute__((ext_vector_type(8)))  float  v8f;
typedef __attribute__((ext_vector_type(4)))  float  v4f;

#define DIN        64          // per-node feature dim
#define TILE_M     128         // edges per block-tile
#define SW1_STRIDE 136         // K=128 + 8 pad (bf16 elems, 272B row = 16B aligned)
#define SW2_STRIDE 264         // K=256 + 8 pad (528B row = 16B aligned)
#define SH1_STRIDE 264

// LDS layout (bytes, all 16B aligned)
#define OFF_W1   0                       // [256][136] bf16 = 69632
#define OFF_W2   69632                   // [256][264] bf16 = 135168
#define OFF_H1   204800                  // [128][264] bf16 = 67584
#define OFF_B1   272384                  // 256 f32
#define OFF_B2   273408                  // 256 f32
#define OFF_W3   274432                  // 256 f32
#define LDS_BYTES 275456

union FragU { v16bf v; v8bf h[2]; };

__global__ __launch_bounds__(256, 1)
void mlp_edge_kernel(const float* __restrict__ xu, const float* __restrict__ xm,
                     const long long* __restrict__ eidx,
                     const float* __restrict__ W1, const float* __restrict__ b1,
                     const float* __restrict__ W2, const float* __restrict__ b2,
                     const float* __restrict__ W3, const float* __restrict__ b3,
                     float* __restrict__ out, int E, int numTiles)
{
    __shared__ __align__(16) unsigned char lds[LDS_BYTES];
    __bf16* sW1 = (__bf16*)(lds + OFF_W1);
    __bf16* sW2 = (__bf16*)(lds + OFF_W2);
    __bf16* sH1 = (__bf16*)(lds + OFF_H1);
    float*  sB1 = (float*)(lds + OFF_B1);
    float*  sB2 = (float*)(lds + OFF_B2);
    float*  sW3 = (float*)(lds + OFF_W3);

    const int tid   = threadIdx.x;
    const int wave  = tid >> 5;
    const int lane  = tid & 31;
    const int lhalf = (lane >> 4);   // 0 or 1
    const int l16   = lane & 15;

    // ---- one-time: biases + transposed bf16 weights into LDS ----
    sB1[tid] = b1[tid];
    sB2[tid] = b2[tid];
    sW3[tid] = W3[tid];
    for (int idx = tid; idx < 128 * 256; idx += 256) {    // W1[k][n] -> sW1[n][k]
        int k = idx >> 8, n = idx & 255;
        sW1[n * SW1_STRIDE + k] = (__bf16)W1[idx];
    }
    for (int idx = tid; idx < 256 * 256; idx += 256) {    // W2[k][n] -> sW2[n][k]
        int k = idx >> 8, n = idx & 255;
        sW2[n * SW2_STRIDE + k] = (__bf16)W2[idx];
    }
    __syncthreads();

    const float bias3 = b3[0];
    const long long* rowIdx = eidx;
    const long long* colIdx = eidx + E;

    const int m0  = wave * 16;        // this wave's 16 rows within the tile
    const int myM = m0 + l16;         // row this lane gathers (A layout: lane = M)

    for (int tile = blockIdx.x; tile < numTiles; tile += gridDim.x) {
        const int e0 = tile * TILE_M;

        // ---- gather edge features straight into layer-1 A fragments ----
        int e  = e0 + myM;
        int ec = (e < E) ? e : 0;
        const float* pu = xu + (size_t)rowIdx[ec] * DIN;
        const float* pm = xm + (size_t)colIdx[ec] * DIN;

        auto loadChunk = [&](int k) -> v8bf {   // 8 consecutive feats at k (k % 8 == 0)
            const float* p = (k < DIN) ? (pu + k) : (pm + (k - DIN));
            v4f f0 = *(const v4f*)p;
            v4f f1 = *(const v4f*)(p + 4);
            v8bf r;
            r[0] = (__bf16)f0[0]; r[1] = (__bf16)f0[1];
            r[2] = (__bf16)f0[2]; r[3] = (__bf16)f0[3];
            r[4] = (__bf16)f1[0]; r[5] = (__bf16)f1[1];
            r[6] = (__bf16)f1[2]; r[7] = (__bf16)f1[3];
            return r;
        };

        v16bf a1[4];
        #pragma unroll
        for (int t = 0; t < 4; ++t) {           // A 16x32: lane-half K split 0..7 / 8..15
            int kA = t * 32 + lhalf * 8;
            FragU fa;
            fa.h[0] = loadChunk(kA);
            fa.h[1] = loadChunk(kA + 16);
            a1[t] = fa.v;
        }

        // ---- layer 1: H1 = relu(X @ W1 + b1), written bf16 to LDS ----
        for (int nt = 0; nt < 16; ++nt) {
            float bias = sB1[nt * 16 + l16];
            v8f acc = {bias, bias, bias, bias, bias, bias, bias, bias};
            #pragma unroll
            for (int kt = 0; kt < 4; ++kt) {
                const __bf16* bp = sW1 + (nt * 16 + l16) * SW1_STRIDE + kt * 32 + lhalf * 8;
                FragU fb;
                fb.h[0] = *(const v8bf*)bp;
                fb.h[1] = *(const v8bf*)(bp + 16);
                acc = __builtin_amdgcn_wmma_f32_16x16x32_bf16(
                        false, a1[kt], false, fb.v, (short)0, acc, false, false);
            }
            int ncol = nt * 16 + l16;           // C layout: lane = N, VGPR r = row r / r+8
            #pragma unroll
            for (int r = 0; r < 8; ++r) {
                float v = acc[r];
                v = v > 0.f ? v : 0.f;
                sH1[(m0 + r + lhalf * 8) * SH1_STRIDE + ncol] = (__bf16)v;
            }
        }

        // ---- hoist layer-2 A fragments (this wave's own H1 rows) ----
        v16bf a2[8];
        #pragma unroll
        for (int kt = 0; kt < 8; ++kt) {
            const __bf16* ap = sH1 + (m0 + l16) * SH1_STRIDE + kt * 32 + lhalf * 8;
            FragU fa;
            fa.h[0] = *(const v8bf*)ap;
            fa.h[1] = *(const v8bf*)(ap + 16);
            a2[kt] = fa.v;
        }

        // ---- layer 2 + folded layer 3: p += relu(H1@W2+b2) * w3 ----
        v8f p = {0.f, 0.f, 0.f, 0.f, 0.f, 0.f, 0.f, 0.f};
        for (int nt = 0; nt < 16; ++nt) {
            float bias = sB2[nt * 16 + l16];
            v8f acc = {bias, bias, bias, bias, bias, bias, bias, bias};
            #pragma unroll
            for (int kt = 0; kt < 8; ++kt) {
                const __bf16* bp = sW2 + (nt * 16 + l16) * SW2_STRIDE + kt * 32 + lhalf * 8;
                FragU fb;
                fb.h[0] = *(const v8bf*)bp;
                fb.h[1] = *(const v8bf*)(bp + 16);
                acc = __builtin_amdgcn_wmma_f32_16x16x32_bf16(
                        false, a2[kt], false, fb.v, (short)0, acc, false, false);
            }
            float w3v = sW3[nt * 16 + l16];
            #pragma unroll
            for (int r = 0; r < 8; ++r) {
                float v = acc[r];
                v = v > 0.f ? v : 0.f;
                p[r] = fmaf(v, w3v, p[r]);
            }
        }

        // ---- reduce 16 partial columns per row, store ----
        #pragma unroll
        for (int r = 0; r < 8; ++r) {
            float s = p[r];
            s += __shfl_xor(s, 1);
            s += __shfl_xor(s, 2);
            s += __shfl_xor(s, 4);
            s += __shfl_xor(s, 8);
            int eo = e0 + m0 + r + lhalf * 8;
            if (l16 == 0 && eo < E) out[eo] = s + bias3;
        }
    }
}

extern "C" void kernel_launch(void* const* d_in, const int* in_sizes, int n_in,
                              void* d_out, int out_size, void* d_ws, size_t ws_size,
                              hipStream_t stream) {
    const float*     xu   = (const float*)d_in[0];
    const float*     xm   = (const float*)d_in[1];
    const long long* eidx = (const long long*)d_in[2];
    const float*     W1   = (const float*)d_in[3];
    const float*     b1   = (const float*)d_in[4];
    const float*     W2   = (const float*)d_in[5];
    const float*     b2   = (const float*)d_in[6];
    const float*     W3   = (const float*)d_in[7];
    const float*     b3   = (const float*)d_in[8];
    float*           out  = (float*)d_out;

    int E        = in_sizes[2] / 2;
    int numTiles = (E + TILE_M - 1) / TILE_M;
    int grid     = numTiles < 512 ? numTiles : 512;

    hipLaunchKernelGGL(mlp_edge_kernel, dim3(grid), dim3(256), 0, stream,
                       xu, xm, eidx, W1, b1, W2, b2, W3, b3, out, E, numTiles);
}
